// VectorQuantizer_79482664779780
// MI455X (gfx1250) — matrix operand order repaired and checked by reading, not compiled, not run on hardware
//
#include <hip/hip_runtime.h>
#include <hip/hip_bf16.h>

// ---------------------------------------------------------------------------
// VQ-VAE vector quantizer for gfx1250 (MI455X), wave32 + WMMA bf16 split-precision
//   z:     (65536, 256) f32      d_in[0]
//   mask:  (65536,)     i32      d_in[1]
//   emb_w: (1024, 256)  f32      d_in[2]
// outputs (f32, concatenated):
//   loss(1) | z_q(16777216) | perplexity(1) | one_hot(67108864) | idx(65536)
//
// GEMM s = z . emb^T done as 3x bf16 WMMA split precision (hi*hi + hi*lo + lo*hi,
// ~2^-17 relative error) on v_wmma_f32_16x16x32_bf16. A-fragments gathered
// directly from global (full 128B line use). emb chunks async-double-buffered
// into LDS with global_load_async_to_lds_b128 when workspace allows a
// pre-converted bf16 hi/lo copy of the codebook.
// ---------------------------------------------------------------------------

#define NROWS   65536
#define DDIM    256
#define NE      1024
#define RPB     128          // rows per block (8 waves x 16 rows)
#define THREADS 256

static const size_t OUT_LOSS   = 0;
static const size_t OUT_ZQ     = 1;
static const size_t OUT_PERP   = 16777217;  // 1 + 128*512*256
static const size_t OUT_ONEHOT = 16777218;
static const size_t OUT_IDX    = 83886082;  // 16777218 + 65536*1024

// ws layout: [0..1023] counts | [1024] sum_sq | [1025] sum_mask | [1026..2049] e_norms
// then (16B aligned) emb_hi bf16[1024*256] @ byte 16384, emb_lo @ byte 540672
#define WS_SUMSQ     1024
#define WS_SUMM      1025
#define WS_ENORM     1026
#define WS_EMBHI_B   16384
#define WS_EMBLO_B   (16384 + 524288)
#define WS_NEED      (16384 + 2*524288)

typedef __attribute__((ext_vector_type(16))) __bf16 v16bf;
typedef __attribute__((ext_vector_type(8)))  float  v8f;
typedef __attribute__((ext_vector_type(2)))  float  v2f;
typedef __attribute__((ext_vector_type(4)))  unsigned v4u;

union FragBF { v16bf v; unsigned u[8]; v4u q[2]; };

// LDS: emb double buffer [2 x (hi 8KB + lo 8KB)] | enorm 4KB | rsum 1KB | idx 512B
#define EBUF_STRIDE 16384
#define ENORM_OFF   32768
#define RSUM_OFF    36864
#define IDX_OFF     37888
#define SMEM_BYTES  38400

static __device__ __forceinline__ unsigned pack2(__bf16 a, __bf16 b) {
    union { __bf16 h[2]; unsigned u; } t;
    t.h[0] = a; t.h[1] = b;
    return t.u;
}
// convert (a,b) f32 -> packed bf16 hi pair + packed bf16 lo (residual) pair
static __device__ __forceinline__ void cvt2(float a, float b, unsigned& uh, unsigned& ul) {
    __bf16 ha = (__bf16)a, hb = (__bf16)b;
    uh = pack2(ha, hb);
    ul = pack2((__bf16)(a - (float)ha), (__bf16)(b - (float)hb));
}

// ---------------------------------------------------------------------------
// prep: per-code ||e||^2, and (if pre) bf16 hi/lo copy of the codebook
__global__ void vq_prep_kernel(const float* __restrict__ emb, float* __restrict__ ws, int pre) {
    int gid = blockIdx.x * blockDim.x + threadIdx.x;
    if (gid < NE) {
        __bf16* hi = (__bf16*)((char*)ws + WS_EMBHI_B);
        __bf16* lo = (__bf16*)((char*)ws + WS_EMBLO_B);
        const float* r = emb + (size_t)gid * DDIM;
        float s = 0.f;
        #pragma unroll 4
        for (int k = 0; k < DDIM; ++k) {
            float v = r[k];
            s += v * v;
            if (pre) {
                __bf16 h = (__bf16)v;
                hi[(size_t)gid * DDIM + k] = h;
                lo[(size_t)gid * DDIM + k] = (__bf16)(v - (float)h);
            }
        }
        ws[WS_ENORM + gid] = s;
    }
}

// ---------------------------------------------------------------------------
__global__ void __launch_bounds__(THREADS)
vq_main_kernel(const float* __restrict__ z, const int* __restrict__ mask,
               const float* __restrict__ emb, float* __restrict__ ws,
               float* __restrict__ out, int pre) {
    extern __shared__ char smem[];
    float* enorm_s = (float*)(smem + ENORM_OFF);
    float* rsum    = (float*)(smem + RSUM_OFF);
    int*   idx_s   = (int*)(smem + IDX_OFF);

    const int t    = threadIdx.x;
    const int lane = t & 31;
    const int wave = t >> 5;
    const int row0 = blockIdx.x * RPB;
    const int m16  = lane & 15;
    const int hh   = lane >> 4;
    const int rw   = wave * 16;
    const int n16  = lane & 15;

    const char* embhi = (const char*)ws + WS_EMBHI_B;   // bf16[1024*256]
    const char* emblo = (const char*)ws + WS_EMBLO_B;
    const unsigned voff = (unsigned)(t * 16);           // 256 thr x 16B = 4KB sweep

    // ---- kick off async staging of chunk 0 ASAP (overlaps A-fragment build)
    if (pre) {
        unsigned ldsH = voff;          // buf0 hi @ 0
        unsigned ldsL = 8192 + voff;   // buf0 lo @ 8192
        const void* sH = embhi;        // chunk 0
        const void* sL = emblo;
        asm volatile(
            "global_load_async_to_lds_b128 %0, %2, %3\n\t"
            "global_load_async_to_lds_b128 %0, %2, %3 offset:4096\n\t"
            "global_load_async_to_lds_b128 %1, %2, %4\n\t"
            "global_load_async_to_lds_b128 %1, %2, %4 offset:4096"
            :: "v"(ldsH), "v"(ldsL), "v"(voff), "s"(sH), "s"(sL) : "memory");
    }

    // ---- stage e-norms
    for (int i = t; i < NE; i += THREADS) enorm_s[i] = ws[WS_ENORM + i];

    // ---- build all A fragments (16-bit A 16x32 layout) straight from global z.
    // lane reads 32B spans; the two lane-halves of a row cover a full 128B line.
    FragBF a_hi[8], a_lo[8];
    {
        const float* zrow = z + (size_t)(row0 + rw + m16) * DDIM;
        #pragma unroll
        for (int kc = 0; kc < 8; ++kc) {
            const int k0 = kc * 32 + 8 * hh;        // K = kc*32 + 8*hh + [0..7]
            const int k1 = kc * 32 + 16 + 8 * hh;   // K = kc*32 + 16 + 8*hh + [0..7]
            float4 f0 = *(const float4*)(zrow + k0);
            float4 f1 = *(const float4*)(zrow + k0 + 4);
            float4 f2 = *(const float4*)(zrow + k1);
            float4 f3 = *(const float4*)(zrow + k1 + 4);
            cvt2(f0.x, f0.y, a_hi[kc].u[0], a_lo[kc].u[0]);
            cvt2(f0.z, f0.w, a_hi[kc].u[1], a_lo[kc].u[1]);
            cvt2(f1.x, f1.y, a_hi[kc].u[2], a_lo[kc].u[2]);
            cvt2(f1.z, f1.w, a_hi[kc].u[3], a_lo[kc].u[3]);
            cvt2(f2.x, f2.y, a_hi[kc].u[4], a_lo[kc].u[4]);
            cvt2(f2.z, f2.w, a_hi[kc].u[5], a_lo[kc].u[5]);
            cvt2(f3.x, f3.y, a_hi[kc].u[6], a_lo[kc].u[6]);
            cvt2(f3.z, f3.w, a_hi[kc].u[7], a_lo[kc].u[7]);
        }
    }

    float minv[8];
    int   mini[8];
    #pragma unroll
    for (int v = 0; v < 8; ++v) { minv[v] = 3.4e38f; mini[v] = 0; }

    // ---- scan all 1024 codes in 16-code chunks, double-buffered in LDS
    for (int cc = 0; cc < NE / 16; ++cc) {
        const int cbase = cc * 16;
        const int buf   = cc & 1;

        if (pre) {
            asm volatile("s_wait_asynccnt 0" ::: "memory");
            __syncthreads();  // current buffer complete for all waves
            if (cc + 1 < NE / 16) {   // prefetch next chunk into other buffer
                const int nb = buf ^ 1;
                unsigned ldsH = (unsigned)(nb * EBUF_STRIDE) + voff;
                unsigned ldsL = ldsH + 8192;
                const void* sH = embhi + (size_t)(cc + 1) * 8192;
                const void* sL = emblo + (size_t)(cc + 1) * 8192;
                asm volatile(
                    "global_load_async_to_lds_b128 %0, %2, %3\n\t"
                    "global_load_async_to_lds_b128 %0, %2, %3 offset:4096\n\t"
                    "global_load_async_to_lds_b128 %1, %2, %4\n\t"
                    "global_load_async_to_lds_b128 %1, %2, %4 offset:4096"
                    :: "v"(ldsH), "v"(ldsL), "v"(voff), "s"(sH), "s"(sL) : "memory");
            }
        } else {
            // fallback: synchronous stage + convert from f32 codebook
            __syncthreads();
            __bf16* ehp = (__bf16*)(smem + buf * EBUF_STRIDE);
            __bf16* elp = ehp + 4096;
            for (int i = t; i < 16 * DDIM; i += THREADS) {
                float f = emb[(size_t)cbase * DDIM + i];
                __bf16 h = (__bf16)f;
                ehp[i] = h;
                elp[i] = (__bf16)(f - (float)h);
            }
            __syncthreads();
        }

        // ---- 16x16x256 tile: 8 k-chunks x 3 split WMMAs, f32 accumulate
        v8f c8 = {0.f, 0.f, 0.f, 0.f, 0.f, 0.f, 0.f, 0.f};
        const v4u* ehq = (const v4u*)(smem + buf * EBUF_STRIDE);
        const v4u* elq = (const v4u*)(smem + buf * EBUF_STRIDE + 8192);
        const int bq = n16 * 32 + hh;          // uint4 idx: B 32x16, K = 8*hh + [0..7], 16 + 8*hh + [0..7]
        #pragma unroll
        for (int kc = 0; kc < 8; ++kc) {
            FragBF bh, bl;
            bh.q[0] = ehq[bq + kc * 4];
            bh.q[1] = ehq[bq + kc * 4 + 2];
            bl.q[0] = elq[bq + kc * 4];
            bl.q[1] = elq[bq + kc * 4 + 2];
            c8 = __builtin_amdgcn_wmma_f32_16x16x32_bf16(false, a_hi[kc].v, false, bh.v, (short)0, c8, false, false);
            c8 = __builtin_amdgcn_wmma_f32_16x16x32_bf16(false, a_hi[kc].v, false, bl.v, (short)0, c8, false, false);
            c8 = __builtin_amdgcn_wmma_f32_16x16x32_bf16(false, a_lo[kc].v, false, bh.v, (short)0, c8, false, false);
        }

        const float en    = enorm_s[cbase + n16];
        const int   myidx = cbase + n16;
        #pragma unroll
        for (int v = 0; v < 8; ++v) {
            float sc = en - 2.0f * c8[v];     // ||e||^2 - 2 z.e  (||z||^2 const per row)
            bool take = sc < minv[v];
            minv[v] = take ? sc : minv[v];
            mini[v] = take ? myidx : mini[v];
        }
    }

    // ---- argmin across the 16 columns (xor masks stay inside each 16-lane half)
    #pragma unroll
    for (int v = 0; v < 8; ++v) {
        #pragma unroll
        for (int off = 1; off < 16; off <<= 1) {
            float ov = __shfl_xor(minv[v], off, 32);
            int   oi = __shfl_xor(mini[v], off, 32);
            if (ov < minv[v] || (ov == minv[v] && oi < mini[v])) { minv[v] = ov; mini[v] = oi; }
        }
        if (n16 == 0) idx_s[rw + 8 * hh + v] = mini[v];   // C layout: row = v + 8*half
    }
    __syncthreads();

    // ---- one-hot: 256MB stream, b64 non-temporal stores (region only 8B aligned)
    {
        size_t base = OUT_ONEHOT + (size_t)row0 * NE;
        for (int i2 = t; i2 < RPB * NE / 2; i2 += THREADS) {
            int i = i2 * 2;
            int r = i >> 10, c = i & (NE - 1);
            int gi = idx_s[r];
            v2f val = {(c == gi) ? 1.0f : 0.0f, (c + 1 == gi) ? 1.0f : 0.0f};
            __builtin_nontemporal_store(val, (v2f*)(out + base + i));
        }
    }

    // ---- z_q gather + masked squared-diff partial sum (scalar NT stores: 4B align)
    float lsum = 0.f;
    {
        size_t zb = (size_t)row0 * DDIM;
        for (int i = t; i < RPB * DDIM; i += THREADS) {
            int r = i >> 8, c = i & (DDIM - 1);
            int gi = idx_s[r];
            float zq = emb[(size_t)gi * DDIM + c];
            __builtin_nontemporal_store(zq, out + OUT_ZQ + zb + i);  // STE fwd == z_q
            float d = zq - z[zb + i];
            lsum += mask[row0 + r] ? d * d : 0.f;
        }
    }
    float msum = (t < RPB) ? (float)mask[row0 + t] : 0.f;

    rsum[t] = lsum; __syncthreads();
    for (int s = 128; s > 0; s >>= 1) { if (t < s) rsum[t] += rsum[t + s]; __syncthreads(); }
    if (t == 0) atomicAdd(&ws[WS_SUMSQ], rsum[0]);
    __syncthreads();
    rsum[t] = msum; __syncthreads();
    for (int s = 128; s > 0; s >>= 1) { if (t < s) rsum[t] += rsum[t + s]; __syncthreads(); }
    if (t == 0) atomicAdd(&ws[WS_SUMM], rsum[0]);

    if (t < RPB) {
        int gi = idx_s[t];
        atomicAdd(&ws[gi], 1.0f);                       // histogram for perplexity
        out[OUT_IDX + row0 + t] = (float)gi;            // idx[:, None] as float
    }
}

// ---------------------------------------------------------------------------
__global__ void vq_finalize_kernel(const float* __restrict__ ws, float* __restrict__ out) {
    __shared__ float red[THREADS];
    int t = threadIdx.x;
    float local = 0.f;
    for (int c = t; c < NE; c += THREADS) {
        float em = ws[c] * (1.0f / (float)NROWS);
        local += em * logf(em + 1e-10f);
    }
    red[t] = local; __syncthreads();
    for (int s = 128; s > 0; s >>= 1) { if (t < s) red[t] += red[t + s]; __syncthreads(); }
    if (t == 0) {
        // commitment and codebook losses are numerically identical forward
        out[OUT_LOSS] = 1.25f * ws[WS_SUMSQ] / (ws[WS_SUMM] + 1e-6f);
        out[OUT_PERP] = expf(-red[0]);
    }
}

// ---------------------------------------------------------------------------
extern "C" void kernel_launch(void* const* d_in, const int* in_sizes, int n_in,
                              void* d_out, int out_size, void* d_ws, size_t ws_size,
                              hipStream_t stream) {
    (void)in_sizes; (void)n_in; (void)out_size;
    const float* z    = (const float*)d_in[0];
    const int*   mask = (const int*)d_in[1];
    const float* emb  = (const float*)d_in[2];
    float* out = (float*)d_out;
    float* ws  = (float*)d_ws;

    const int pre = (ws_size >= (size_t)WS_NEED) ? 1 : 0;

    // zero counts + accumulators
    (void)hipMemsetAsync(d_ws, 0, WS_ENORM * sizeof(float), stream);

    vq_prep_kernel<<<NE / THREADS, THREADS, 0, stream>>>(emb, ws, pre);
    vq_main_kernel<<<NROWS / RPB, THREADS, SMEM_BYTES, stream>>>(z, mask, emb, ws, out, pre);
    vq_finalize_kernel<<<1, THREADS, 0, stream>>>(ws, out);
}
